// quality_30288109371429
// MI455X (gfx1250) — compile-verified, matrix operand order checked by reading, not run on hardware
//
#include <hip/hip_runtime.h>

// ---------------------------------------------------------------------------
// Problem constants (from the reference)
// ---------------------------------------------------------------------------
constexpr int CB    = 192;          // batch
constexpr int CT    = 50;           // time
constexpr int CM    = CB * CT;      // 9600 rows, divisible by 16
constexpr int CVOC  = 24000;        // vocab (divisible by 64)
constexpr int CE    = 300;          // embed
constexpr int CHID2 = 2000;         // 2*hidden
constexpr int CQV   = 300;          // quality vector / maxout units
constexpr int CH    = 100;          // LSTM hidden

// Padded K dims (multiples of 64 for the 2x-unrolled double-buffered WMMA
// loop) and padded weight-row counts (multiples of 64 so a 16x64-per-wave
// tile never reads OOB rows). Pads are zero-filled.
constexpr int KP_HID2 = 2048;   // 2000 -> 32*64
constexpr int KP_E2   = 640;    // 600  -> 10*64
constexpr int KP_QV   = 320;    // 300  -> 5*64
constexpr int NP_600  = 640;    // 600  -> 64-pad
constexpr int NP_300  = 320;    // 300  -> 64-pad

typedef __attribute__((ext_vector_type(16))) _Float16 v16h;
typedef __attribute__((ext_vector_type(8)))  _Float16 v8h;
typedef __attribute__((ext_vector_type(8)))  float    v8f;

// ---------------------------------------------------------------------------
// f32 -> f16 conversion with zero padding:  src [rows, K] -> dst [rows_p, Kp]
// ---------------------------------------------------------------------------
__global__ void cvt_pad_f16(const float* __restrict__ src, _Float16* __restrict__ dst,
                            long rows, int K, long rows_p, int Kp)
{
    size_t total = (size_t)rows_p * (size_t)Kp;
    for (size_t i = (size_t)blockIdx.x * blockDim.x + threadIdx.x; i < total;
         i += (size_t)gridDim.x * blockDim.x) {
        size_t r = i / (size_t)Kp;
        int    k = (int)(i - r * (size_t)Kp);
        float  v = (r < (size_t)rows && k < K) ? src[r * (size_t)K + k] : 0.0f;
        dst[i] = (_Float16)v;
    }
}

// ---------------------------------------------------------------------------
// tensor2 = concat(embed[y[:,li]], embed[y[:,ri]]) -> f16, padded to Kp cols
// ---------------------------------------------------------------------------
__global__ void build_tensor2_f16(const float* __restrict__ emb, const int* __restrict__ y,
                                  _Float16* __restrict__ dst, int Kp)
{
    size_t total = (size_t)CM * (size_t)Kp;
    for (size_t i = (size_t)blockIdx.x * blockDim.x + threadIdx.x; i < total;
         i += (size_t)gridDim.x * blockDim.x) {
        size_t row = i / (size_t)Kp;
        int    k   = (int)(i - row * (size_t)Kp);
        int    b   = (int)(row / CT);
        int    t   = (int)(row - (size_t)b * CT);
        float  v   = 0.0f;
        if (k < 2 * CE) {
            int src_t = (k < CE) ? ((t == 0) ? 1 : t - 1)
                                 : ((t == CT - 1) ? CT - 2 : t + 1);
            int e     = (k < CE) ? k : k - CE;
            int tok   = y[b * CT + src_t];
            v = emb[(size_t)tok * CE + e];
        }
        dst[i] = (_Float16)v;
    }
}

// ---------------------------------------------------------------------------
// WMMA GEMM:  C[M,N] (f32) (+)= A[M,Kp](f16) * W[N,Kp](f16)^T  + bias
// One wave -> 16(M) x 64(N) tile (4 accumulators, A fragment reused 4x).
// Fragment layout per ISA 7.12.2 (16-bit, wave32): lanes 0-15 hold
// K {0..7,16..23}; lanes 16-31 hold K {8..15,24..31}.
// K loop: 2x unrolled with two STATIC fragment sets (A/B). Loads for the
// next half-block issue before the WMMAs of the current one, and because
// the buffers are static there are no register rotations (no v_mov chains,
// no WMMA->VALU hazard padding).
// ---------------------------------------------------------------------------
__device__ __forceinline__ v16h load_frag(const _Float16* __restrict__ p, int off)
{
    v8h lo = *(const v8h*)(p + off);
    v8h hi = *(const v8h*)(p + 16 + off);
    v16h r;
#pragma unroll
    for (int i = 0; i < 8; ++i) { r[i] = lo[i]; r[i + 8] = hi[i]; }
    return r;
}

__global__ __launch_bounds__(128) void wmma_gemm_f16(
    const _Float16* __restrict__ A, int lda,
    const _Float16* __restrict__ Bw, int ldb,
    float* __restrict__ C, int ldc,
    int N, int Kp,
    const float* __restrict__ bias, int accumulate)
{
    const int lane = threadIdx.x;                       // wave32
    const int m0   = blockIdx.y << 4;
    const int n0   = (blockIdx.x * 4 + threadIdx.y) << 6;
    if (n0 >= N) return;                                // uniform per wave

    const int half = (lane >= 16) ? 8 : 0;
    const int l15  = lane & 15;

    const _Float16* arow = A + (size_t)(m0 + l15) * lda;
    const _Float16* br0  = Bw + (size_t)(n0 +  0 + l15) * ldb;
    const _Float16* br1  = Bw + (size_t)(n0 + 16 + l15) * ldb;
    const _Float16* br2  = Bw + (size_t)(n0 + 32 + l15) * ldb;
    const _Float16* br3  = Bw + (size_t)(n0 + 48 + l15) * ldb;

    v8f acc0 = {}, acc1 = {}, acc2 = {}, acc3 = {};

    // --- prologue: load K-block 0 into set A
    v16h aA  = load_frag(arow, half);
    v16h bA0 = load_frag(br0,  half);
    v16h bA1 = load_frag(br1,  half);
    v16h bA2 = load_frag(br2,  half);
    v16h bA3 = load_frag(br3,  half);

    for (int k = 0; k < Kp; k += 64) {                  // Kp % 64 == 0
        // issue loads for K-block k+32 into set B (independent of set-A WMMAs)
        v16h aB  = load_frag(arow + k + 32, half);
        v16h bB0 = load_frag(br0  + k + 32, half);
        v16h bB1 = load_frag(br1  + k + 32, half);
        v16h bB2 = load_frag(br2  + k + 32, half);
        v16h bB3 = load_frag(br3  + k + 32, half);
        // near-cache prefetch four blocks ahead (working set is L2-resident)
        __builtin_prefetch(arow + k + 256, 0, 3);
        __builtin_prefetch(br0  + k + 256, 0, 3);
        __builtin_prefetch(br2  + k + 256, 0, 3);

        acc0 = __builtin_amdgcn_wmma_f32_16x16x32_f16(false, aA, false, bA0, (short)0, acc0, false, false);
        acc1 = __builtin_amdgcn_wmma_f32_16x16x32_f16(false, aA, false, bA1, (short)0, acc1, false, false);
        acc2 = __builtin_amdgcn_wmma_f32_16x16x32_f16(false, aA, false, bA2, (short)0, acc2, false, false);
        acc3 = __builtin_amdgcn_wmma_f32_16x16x32_f16(false, aA, false, bA3, (short)0, acc3, false, false);

        if (k + 64 < Kp) {                              // uniform branch
            // refill set A with K-block k+64 while set-B WMMAs run
            aA  = load_frag(arow + k + 64, half);
            bA0 = load_frag(br0  + k + 64, half);
            bA1 = load_frag(br1  + k + 64, half);
            bA2 = load_frag(br2  + k + 64, half);
            bA3 = load_frag(br3  + k + 64, half);
        }

        acc0 = __builtin_amdgcn_wmma_f32_16x16x32_f16(false, aB, false, bB0, (short)0, acc0, false, false);
        acc1 = __builtin_amdgcn_wmma_f32_16x16x32_f16(false, aB, false, bB1, (short)0, acc1, false, false);
        acc2 = __builtin_amdgcn_wmma_f32_16x16x32_f16(false, aB, false, bB2, (short)0, acc2, false, false);
        acc3 = __builtin_amdgcn_wmma_f32_16x16x32_f16(false, aB, false, bB3, (short)0, acc3, false, false);
    }

    // D layout: VGPR r -> row m0+r (lanes 0-15) / m0+r+8 (lanes 16-31), col = n0+l15
    v8f accs[4] = {acc0, acc1, acc2, acc3};
#pragma unroll
    for (int s = 0; s < 4; ++s) {
        int col = n0 + 16 * s + l15;
        if (col < N) {
            float bv = bias ? bias[col] : 0.0f;
#pragma unroll
            for (int r = 0; r < 8; ++r) {
                int    row = m0 + half + r;
                size_t idx = (size_t)row * ldc + col;
                float  v   = accs[s][r] + bv;
                if (accumulate) v += C[idx];
                C[idx] = v;
            }
        }
    }
}

// ---------------------------------------------------------------------------
// maxout pairs + f16 conversion:  t[r,j] = max(t_long[r,2j], t_long[r,2j+1])
// ---------------------------------------------------------------------------
__global__ void maxout_cvt_f16(const float* __restrict__ tl, _Float16* __restrict__ dst,
                               int halfN, int Kp, int ld)
{
    size_t total = (size_t)CM * (size_t)Kp;
    for (size_t i = (size_t)blockIdx.x * blockDim.x + threadIdx.x; i < total;
         i += (size_t)gridDim.x * blockDim.x) {
        size_t r = i / (size_t)Kp;
        int    k = (int)(i - r * (size_t)Kp);
        float  v = 0.0f;
        if (k < halfN) {
            float x0 = tl[r * (size_t)ld + 2 * k];
            float x1 = tl[r * (size_t)ld + 2 * k + 1];
            v = fmaxf(x0, x1);
        }
        dst[i] = (_Float16)v;
    }
}

// ---------------------------------------------------------------------------
// quality[r,j] = W1[y[r], j] * a[r,j]
// ---------------------------------------------------------------------------
__global__ void quality_gather(const float* __restrict__ a, const float* __restrict__ W1,
                               const int* __restrict__ y, float* __restrict__ q)
{
    size_t total = (size_t)CM * (size_t)CQV;
    for (size_t i = (size_t)blockIdx.x * blockDim.x + threadIdx.x; i < total;
         i += (size_t)gridDim.x * blockDim.x) {
        size_t r = i / (size_t)CQV;
        int    k = (int)(i - r * (size_t)CQV);
        int  tok = y[r];
        q[i] = W1[(size_t)tok * CQV + k] * a[i];
    }
}

// ---------------------------------------------------------------------------
// In-place log-softmax over rows of x [CM, ncols]
// ---------------------------------------------------------------------------
__global__ __launch_bounds__(256) void log_softmax_rows(float* __restrict__ x, int ncols)
{
    __shared__ float red[256];
    const int tid  = threadIdx.x;
    float*    row  = x + (size_t)blockIdx.x * ncols;

    float m = -3.4e38f;
    for (int c = tid; c < ncols; c += 256) m = fmaxf(m, row[c]);
    red[tid] = m; __syncthreads();
    for (int s = 128; s > 0; s >>= 1) {
        if (tid < s) red[tid] = fmaxf(red[tid], red[tid + s]);
        __syncthreads();
    }
    m = red[0]; __syncthreads();

    float sum = 0.0f;
    for (int c = tid; c < ncols; c += 256) sum += expf(row[c] - m);
    red[tid] = sum; __syncthreads();
    for (int s = 128; s > 0; s >>= 1) {
        if (tid < s) red[tid] += red[tid + s];
        __syncthreads();
    }
    float lse = m + logf(red[0]);
    __syncthreads();
    for (int c = tid; c < ncols; c += 256) row[c] -= lse;
}

// ---------------------------------------------------------------------------
// Fused 2-layer LSTM + QE head. One workgroup per batch element; the
// recurrence is independent across b, so all 50 steps stay in one kernel
// with h/c state resident in LDS.
// ---------------------------------------------------------------------------
__global__ __launch_bounds__(256) void lstm_qe_kernel(
    const float* __restrict__ quality, const int* __restrict__ y,
    const float* __restrict__ wih0, const float* __restrict__ whh0,
    const float* __restrict__ bih0, const float* __restrict__ bhh0,
    const float* __restrict__ wih1, const float* __restrict__ whh1,
    const float* __restrict__ bih1, const float* __restrict__ bhh1,
    const float* __restrict__ Wqe_w, const float* __restrict__ Wqe_b,
    float* __restrict__ qe_out)
{
    const int b   = blockIdx.x;
    const int tid = threadIdx.x;
    __shared__ float xs[CQV];
    __shared__ float h0s[CH], c0s[CH], h1s[CH], c1s[CH], hsel[CH];
    __shared__ float g0s[4 * CH], g1s[4 * CH];
    __shared__ int   idx1;

    for (int i = tid; i < CH; i += 256) { h0s[i]=0.f; c0s[i]=0.f; h1s[i]=0.f; c1s[i]=0.f; hsel[i]=0.f; }
    if (tid == 0) {
        int f = CT - 1;
        for (int t = 0; t < CT; ++t) if (y[b * CT + t] == 0) { f = t; break; }
        idx1 = f;
    }
    __syncthreads();

    for (int t = 0; t < CT; ++t) {
        for (int i = tid; i < CQV; i += 256)
            xs[i] = quality[((size_t)b * CT + t) * CQV + i];
        __syncthreads();

        // layer 0 gates (torch order i,f,g,o)
        for (int g = tid; g < 4 * CH; g += 256) {
            float s = bih0[g] + bhh0[g];
            const float* wr = wih0 + (size_t)g * CQV;
            for (int k = 0; k < CQV; ++k) s = fmaf(wr[k], xs[k], s);
            const float* hr = whh0 + (size_t)g * CH;
            for (int k = 0; k < CH; ++k) s = fmaf(hr[k], h0s[k], s);
            g0s[g] = s;
        }
        __syncthreads();
        for (int i = tid; i < CH; i += 256) {
            float ig = 1.f / (1.f + expf(-g0s[i]));
            float fg = 1.f / (1.f + expf(-g0s[CH + i]));
            float gg = tanhf(g0s[2 * CH + i]);
            float og = 1.f / (1.f + expf(-g0s[3 * CH + i]));
            float c  = fg * c0s[i] + ig * gg;
            c0s[i] = c;
            h0s[i] = og * tanhf(c);
        }
        __syncthreads();

        // layer 1 gates
        for (int g = tid; g < 4 * CH; g += 256) {
            float s = bih1[g] + bhh1[g];
            const float* wr = wih1 + (size_t)g * CH;
            for (int k = 0; k < CH; ++k) s = fmaf(wr[k], h0s[k], s);
            const float* hr = whh1 + (size_t)g * CH;
            for (int k = 0; k < CH; ++k) s = fmaf(hr[k], h1s[k], s);
            g1s[g] = s;
        }
        __syncthreads();
        for (int i = tid; i < CH; i += 256) {
            float ig = 1.f / (1.f + expf(-g1s[i]));
            float fg = 1.f / (1.f + expf(-g1s[CH + i]));
            float gg = tanhf(g1s[2 * CH + i]);
            float og = 1.f / (1.f + expf(-g1s[3 * CH + i]));
            float c  = fg * c1s[i] + ig * gg;
            c1s[i] = c;
            float h = og * tanhf(c);
            h1s[i] = h;
            if (t == idx1) hsel[i] = h;
        }
        __syncthreads();
    }

    if (tid == 0) {
        float s = Wqe_b[0];
        for (int k = 0; k < CH; ++k) s = fmaf(Wqe_w[k], hsel[k], s);
        qe_out[b] = 1.f / (1.f + expf(-s));
    }
}

// ---------------------------------------------------------------------------
// Host orchestration
// ---------------------------------------------------------------------------
extern "C" void kernel_launch(void* const* d_in, const int* in_sizes, int n_in,
                              void* d_out, int out_size, void* d_ws, size_t ws_size,
                              hipStream_t stream)
{
    (void)in_sizes; (void)n_in; (void)out_size; (void)ws_size;

    const float* dec_out = (const float*)d_in[0];
    const int*   y       = (const int*)  d_in[1];
    const float* contex  = (const float*)d_in[2];
    const float* emb     = (const float*)d_in[3];
    const float* W1      = (const float*)d_in[4];
    const float* S_w     = (const float*)d_in[5];
    const float* S_b     = (const float*)d_in[6];
    const float* V_w     = (const float*)d_in[7];
    const float* V_b     = (const float*)d_in[8];
    const float* C0_w    = (const float*)d_in[9];
    const float* C0_b    = (const float*)d_in[10];
    const float* W2_w    = (const float*)d_in[11];
    const float* W2_b    = (const float*)d_in[12];
    const float* wih0    = (const float*)d_in[13];
    const float* whh0    = (const float*)d_in[14];
    const float* bih0    = (const float*)d_in[15];
    const float* bhh0    = (const float*)d_in[16];
    const float* wih1    = (const float*)d_in[17];
    const float* whh1    = (const float*)d_in[18];
    const float* bih1    = (const float*)d_in[19];
    const float* bhh1    = (const float*)d_in[20];
    const float* Wqe_w   = (const float*)d_in[21];
    const float* Wqe_b   = (const float*)d_in[22];

    float* out = (float*)d_out;           // [CM, CVOC] log_softmax, then [CB] QE

    // Workspace carving (256B aligned)
    char*  ws  = (char*)d_ws;
    size_t off = 0;
    auto carve = [&](size_t bytes) -> char* {
        char* p = ws + off;
        off = (off + bytes + 255) & ~(size_t)255;
        return p;
    };
    _Float16* decf16 = (_Float16*)carve((size_t)CM * KP_HID2 * 2);
    _Float16* ctxf16 = (_Float16*)carve((size_t)CM * KP_HID2 * 2);
    _Float16* t2f16  = (_Float16*)carve((size_t)CM * KP_E2   * 2);
    _Float16* Swf16  = (_Float16*)carve((size_t)NP_600 * KP_HID2 * 2);
    _Float16* C0f16  = (_Float16*)carve((size_t)NP_600 * KP_HID2 * 2);
    _Float16* Vwf16  = (_Float16*)carve((size_t)NP_600 * KP_E2   * 2);
    _Float16* W2f16  = (_Float16*)carve((size_t)NP_300 * KP_QV   * 2);
    _Float16* W1f16  = (_Float16*)carve((size_t)CVOC   * KP_QV   * 2);
    float*    tlong  = (float*)   carve((size_t)CM * 600 * 4);
    _Float16* tf16   = (_Float16*)carve((size_t)CM * KP_QV * 2);
    float*    af32   = (float*)   carve((size_t)CM * CQV * 4);
    _Float16* af16   = (_Float16*)carve((size_t)CM * KP_QV * 2);
    float*    qual   = (float*)   carve((size_t)CM * CQV * 4);

    const int CV = 4096;  // grid-stride blocks for conversion kernels

    // --- Stage 0: one-time f16 conversions (K padded to x64, W rows to x64)
    cvt_pad_f16<<<CV, 256, 0, stream>>>(dec_out, decf16, CM, CHID2, CM, KP_HID2);
    cvt_pad_f16<<<CV, 256, 0, stream>>>(contex,  ctxf16, CM, CHID2, CM, KP_HID2);
    build_tensor2_f16<<<CV, 256, 0, stream>>>(emb, y, t2f16, KP_E2);
    cvt_pad_f16<<<CV, 256, 0, stream>>>(S_w,  Swf16, 600,  CHID2, NP_600, KP_HID2);
    cvt_pad_f16<<<CV, 256, 0, stream>>>(C0_w, C0f16, 600,  CHID2, NP_600, KP_HID2);
    cvt_pad_f16<<<CV, 256, 0, stream>>>(V_w,  Vwf16, 600,  600,   NP_600, KP_E2);
    cvt_pad_f16<<<CV, 256, 0, stream>>>(W2_w, W2f16, 300,  300,   NP_300, KP_QV);
    cvt_pad_f16<<<CV, 256, 0, stream>>>(W1,   W1f16, CVOC, 300,   CVOC,   KP_QV);

    dim3 gblk(32, 4);   // 4 waves/block, each owns a 16x64 C tile
    const int MT = CM / 16;   // 600

    // --- Stage 1: t_long = dec@S^T + tensor2@V^T + contex@C0^T (+ biases)
    wmma_gemm_f16<<<dim3((600 + 255) / 256, MT), gblk, 0, stream>>>(
        decf16, KP_HID2, Swf16, KP_HID2, tlong, 600, 600, KP_HID2, S_b, 0);
    wmma_gemm_f16<<<dim3((600 + 255) / 256, MT), gblk, 0, stream>>>(
        t2f16, KP_E2, Vwf16, KP_E2, tlong, 600, 600, KP_E2, V_b, 1);
    wmma_gemm_f16<<<dim3((600 + 255) / 256, MT), gblk, 0, stream>>>(
        ctxf16, KP_HID2, C0f16, KP_HID2, tlong, 600, 600, KP_HID2, C0_b, 1);

    // --- Stage 2: maxout -> t (f16, padded), then a = t @ W2^T + W2_b
    maxout_cvt_f16<<<CV, 256, 0, stream>>>(tlong, tf16, CQV, KP_QV, 600);
    wmma_gemm_f16<<<dim3((300 + 255) / 256, MT), gblk, 0, stream>>>(
        tf16, KP_QV, W2f16, KP_QV, af32, CQV, 300, KP_QV, W2_b, 0);
    cvt_pad_f16<<<CV, 256, 0, stream>>>(af32, af16, CM, CQV, CM, KP_QV);

    // --- Stage 3: logits = a @ W1^T  (written straight into d_out)
    wmma_gemm_f16<<<dim3((CVOC + 255) / 256, MT), gblk, 0, stream>>>(
        af16, KP_QV, W1f16, KP_QV, out, CVOC, CVOC, KP_QV, nullptr, 0);

    // --- Stage 4: in-place log-softmax over vocab rows
    log_softmax_rows<<<CM, 256, 0, stream>>>(out, CVOC);

    // --- Stage 5: quality = W1[y] * a ; fused 2-layer LSTM + QE head
    quality_gather<<<CV, 256, 0, stream>>>(af32, W1, y, qual);
    lstm_qe_kernel<<<CB, 256, 0, stream>>>(
        qual, y, wih0, whh0, bih0, bhh0, wih1, whh1, bih1, bhh1,
        Wqe_w, Wqe_b, out + (size_t)CM * CVOC);
}
